// Attention_35416300323707
// MI455X (gfx1250) — matrix operand order repaired and checked by reading, not compiled, not run on hardware
//
#include <hip/hip_runtime.h>
#include <hip/hip_bf16.h>

#define BATCH  2
#define NSEQ   2048
#define DMODEL 1024
#define NHEAD  16
#define HDIM   64
#define HND    (NHEAD * HDIM)   // 1024

typedef __attribute__((ext_vector_type(16))) _Float16 v16h;
typedef __attribute__((ext_vector_type(8)))  _Float16 h8;
typedef __attribute__((ext_vector_type(4)))  _Float16 h4;
typedef __attribute__((ext_vector_type(8)))  float    v8f;

__device__ __forceinline__ v16h cat16(h8 a, h8 b) {
  v16h r;
#pragma unroll
  for (int i = 0; i < 8; ++i) { r[i] = a[i]; r[i + 8] = b[i]; }
  return r;
}

__device__ __forceinline__ v8f wmma_f16(v16h a, v16h b, v8f c) {
  // D = A(16x32 f16) * B(32x16 f16) + C(16x16 f32)
  return __builtin_amdgcn_wmma_f32_16x16x32_f16(
      /*neg_a=*/false, a, /*neg_b=*/false, b,
      /*c_mod=*/(short)0, c, /*reuse_a=*/false, /*reuse_b=*/false);
}

// ---------------------------------------------------------------------------
// Kernel 1: fp32 -> fp16 conversion (x)
// ---------------------------------------------------------------------------
__global__ void cvt_f32_f16(const float* __restrict__ src,
                            _Float16* __restrict__ dst, int n4) {
  int i = blockIdx.x * blockDim.x + threadIdx.x;
  int stride = gridDim.x * blockDim.x;
  for (; i < n4; i += stride) {
    float4 v = ((const float4*)src)[i];
    h4 o;
    o[0] = (_Float16)v.x; o[1] = (_Float16)v.y;
    o[2] = (_Float16)v.z; o[3] = (_Float16)v.w;
    ((h4*)dst)[i] = o;
  }
}

// ---------------------------------------------------------------------------
// Kernel 2: QKV projection (WMMA) + bias + RMSNorm + RoPE epilogue.
// Block = 256 thr (8 waves). One block: one (b, head, 64-row n strip).
// Wave w: row group (w&3) of 16 rows, col groups {(w>>2), (w>>2)+2} of 16 hd.
// q is pre-scaled by log2(e)/sqrt(HD) so attention softmax runs in exp2.
// ---------------------------------------------------------------------------
__global__ __launch_bounds__(256) void qkv_rope_kernel(
    const _Float16* __restrict__ xh,
    const float* __restrict__ Wq, const float* __restrict__ bq,
    const float* __restrict__ Wk, const float* __restrict__ bk,
    const float* __restrict__ Wv, const float* __restrict__ bv,
    const float* __restrict__ qn_w, const float* __restrict__ kn_w,
    const float* __restrict__ freqs,
    _Float16* __restrict__ qf, _Float16* __restrict__ kf,
    _Float16* __restrict__ vf) {
  __shared__ __align__(32) char smem[17408];
  _Float16* ldsW = (_Float16*)smem;         // 3 * (64 col * 32 k) f16 = 12 KB
  float*    tile = (float*)smem;            // 64x64 f32 = 16 KB (epilogue)
  float*    part = (float*)(smem + 16384);  // 256 f32 partial sums

  const int tid  = threadIdx.x;
  const int wave = tid >> 5, lane = tid & 31;
  const int half = lane >> 4, ln = lane & 15;
  const int nblk = NSEQ / 64;
  const int b  = blockIdx.x / (NHEAD * nblk);
  const int h  = (blockIdx.x / nblk) % NHEAD;
  const int n0 = (blockIdx.x % nblk) * 64;

  const int rowg = wave & 3;
  const int cg0  = wave >> 2;
  const int arow = n0 + rowg * 16 + ln;

  v8f zero = {};
  v8f acc[3][2];
#pragma unroll
  for (int m = 0; m < 3; ++m) { acc[m][0] = zero; acc[m][1] = zero; }

  for (int kk = 0; kk < DMODEL; kk += 32) {
    __syncthreads();
    // Stage W tiles (32k x 64n) into LDS column-major: col*32 + k.
#pragma unroll
    for (int mat = 0; mat < 3; ++mat) {
      const float* W = (mat == 0) ? Wq : (mat == 1) ? Wk : Wv;
      _Float16* dst = ldsW + mat * 2048;
#pragma unroll
      for (int it = 0; it < 2; ++it) {
        int idx = tid * 2 + it;            // 0..511 float4 slots
        int k   = idx >> 4;                // 0..31
        int n4  = (idx & 15) * 4;          // 0..60
        float4 w4 = *(const float4*)(W + (size_t)(kk + k) * HND + h * HDIM + n4);
        dst[(n4 + 0) * 32 + k] = (_Float16)w4.x;
        dst[(n4 + 1) * 32 + k] = (_Float16)w4.y;
        dst[(n4 + 2) * 32 + k] = (_Float16)w4.z;
        dst[(n4 + 3) * 32 + k] = (_Float16)w4.w;
      }
    }
    __syncthreads();
    // A fragment (x row strip) directly from global f16.
    const _Float16* ab = xh + ((size_t)b * NSEQ + arow) * DMODEL + kk + 16 * half;
    v16h afrag = cat16(*(const h8*)ab, *(const h8*)(ab + 8));
#pragma unroll
    for (int mat = 0; mat < 3; ++mat) {
#pragma unroll
      for (int cgi = 0; cgi < 2; ++cgi) {
        int col = (cg0 + cgi * 2) * 16 + ln;
        const _Float16* bb = ldsW + mat * 2048 + col * 32 + 16 * half;
        v16h bfrag = *(const v16h*)bb;      // 32 contiguous bytes
        acc[mat][cgi] = wmma_f16(afrag, bfrag, acc[mat][cgi]);
      }
    }
  }

  // Epilogue: bias + RMSNorm + RoPE via LDS staging, one matrix at a time.
  const int row   = tid >> 2;  // 0..63
  const int qq    = tid & 3;   // 16-col chunk
  const int nglob = n0 + row;
#pragma unroll
  for (int mat = 0; mat < 3; ++mat) {
    __syncthreads();
#pragma unroll
    for (int cgi = 0; cgi < 2; ++cgi) {
      int col   = (cg0 + cgi * 2) * 16 + ln;
      int rbase = rowg * 16 + 8 * half;
#pragma unroll
      for (int r = 0; r < 8; ++r)
        tile[(rbase + r) * 64 + col] = acc[mat][cgi][r];
    }
    __syncthreads();
    const float* bias = (mat == 0) ? bq : (mat == 1) ? bk : bv;
    float v[16];
    float ss = 0.f;
#pragma unroll
    for (int i = 0; i < 16; ++i) {
      int c = qq * 16 + i;
      v[i] = tile[row * 64 + c] + bias[h * HDIM + c];
      ss += v[i] * v[i];
    }
    part[tid] = ss;
    __syncthreads();
    size_t obase = ((size_t)(b * NHEAD + h) * NSEQ + nglob) * HDIM;
    if (mat < 2) {
      float s = part[row * 4] + part[row * 4 + 1] +
                part[row * 4 + 2] + part[row * 4 + 3];
      float rs = rsqrtf(s * (1.f / HDIM) + 1e-6f);
      const float* nw = (mat == 0) ? qn_w : kn_w;
      _Float16* op = (mat == 0) ? qf : kf;
      // fold 1/sqrt(HD) * log2(e) into q so softmax can use exp2 directly
      float oscale = (mat == 0) ? (0.125f * 1.44269504088896f) : 1.f;
#pragma unroll
      for (int i = 0; i < 16; i += 2) {
        int c0 = qq * 16 + i, c1 = c0 + 1;
        float y0 = v[i] * rs * nw[c0];
        float y1 = v[i + 1] * rs * nw[c1];
        float f0 = freqs[(size_t)nglob * HDIM + c0];
        float f1 = freqs[(size_t)nglob * HDIM + c1];
        // out = t*cos + rotate_half(t)*sin, rh = (-x2, x1)
        float o0 = (y0 * __cosf(f0) - y1 * __sinf(f0)) * oscale;
        float o1 = (y1 * __cosf(f1) + y0 * __sinf(f1)) * oscale;
        op[obase + c0] = (_Float16)o0;
        op[obase + c1] = (_Float16)o1;
      }
    } else {
#pragma unroll
      for (int i = 0; i < 16; ++i)
        vf[obase + qq * 16 + i] = (_Float16)v[i];
    }
  }
}

// ---------------------------------------------------------------------------
// Kernel 3: flash attention, one (b,h,128 q rows) per block; each of 8 waves
// owns 16 q rows independently. 64-key tiles amortize the per-tile softmax
// VALU (shfl reductions, running stats, O rescale) over twice the keys.
// Scores are in log2 units (q pre-scaled by log2e/sqrt(HD)) -> exp2f.
// ---------------------------------------------------------------------------
__global__ __launch_bounds__(256) void flash_attn_kernel(
    const _Float16* __restrict__ qf, const _Float16* __restrict__ kf,
    const _Float16* __restrict__ vf, const unsigned char* __restrict__ maskp,
    _Float16* __restrict__ attnb) {
  __shared__ __align__(32) _Float16 ldsV[HDIM * 64];     // [hd][key], 8 KB
  __shared__ __align__(32) _Float16 ldsP[8 * 16 * 64];   // per-wave P, 16 KB

  const int tid  = threadIdx.x;
  const int wave = tid >> 5, lane = tid & 31;
  const int half = lane >> 4, ln = lane & 15;
  const int qblocks = NSEQ / 128;
  const int b  = blockIdx.x / (NHEAD * qblocks);
  const int h  = (blockIdx.x / qblocks) % NHEAD;
  const int qb = blockIdx.x % qblocks;
  const int qrow0 = qb * 128 + wave * 16;

  const size_t headoff = (size_t)(b * NHEAD + h) * NSEQ * HDIM;
  const _Float16* qbase = qf + headoff + (size_t)(qrow0 + ln) * HDIM;
  v16h qa0 = cat16(*(const h8*)(qbase + 16 * half),
                   *(const h8*)(qbase + 8 + 16 * half));
  v16h qa1 = cat16(*(const h8*)(qbase + 32 + 16 * half),
                   *(const h8*)(qbase + 40 + 16 * half));

  v8f zero = {};
  v8f o[4];
#pragma unroll
  for (int c = 0; c < 4; ++c) o[c] = zero;
  float mrun[8], lrun[8];
#pragma unroll
  for (int r = 0; r < 8; ++r) { mrun[r] = -1e30f; lrun[r] = 0.f; }

  const _Float16* kb = kf + headoff;
  const _Float16* vb = vf + headoff;
  _Float16* pw = ldsP + wave * 1024;

  for (int kt = 0; kt < NSEQ; kt += 64) {
    __syncthreads();
    {  // stage V tile (64 keys x 64 hd) transposed: ldsV[hd*64 + key]
      int key = tid >> 2;            // 0..63
      int c16 = (tid & 3) * 16;      // hd chunk
      v16h vv = *(const v16h*)(vb + (size_t)(kt + key) * HDIM + c16);
#pragma unroll
      for (int j = 0; j < 16; ++j) ldsV[(c16 + j) * 64 + key] = vv[j];
    }
    __syncthreads();

    // scores: s = q (16x64) @ k^T (64x64); K B-fragments direct from global
    v8f s[4];
#pragma unroll
    for (int sub = 0; sub < 4; ++sub) {
      const _Float16* kp = kb + (size_t)(kt + sub * 16 + ln) * HDIM;
      v16h bk0 = *(const v16h*)(kp + 16 * half);
      v16h bk1 = *(const v16h*)(kp + 32 + 16 * half);
      s[sub] = wmma_f16(qa0, bk0, zero);
      s[sub] = wmma_f16(qa1, bk1, s[sub]);
    }
    // key-padding mask (column-wise)
#pragma unroll
    for (int sub = 0; sub < 4; ++sub) {
      if (!maskp[b * NSEQ + kt + sub * 16 + ln]) {
#pragma unroll
        for (int r = 0; r < 8; ++r) s[sub][r] = -1e30f;
      }
    }
    // online softmax per q row (row lives in 16 lanes of one half), log2 dom.
#pragma unroll
    for (int r = 0; r < 8; ++r) {
      float t = fmaxf(fmaxf(s[0][r], s[1][r]), fmaxf(s[2][r], s[3][r]));
      t = fmaxf(t, __shfl_xor(t, 1));
      t = fmaxf(t, __shfl_xor(t, 2));
      t = fmaxf(t, __shfl_xor(t, 4));
      t = fmaxf(t, __shfl_xor(t, 8));
      float nm = fmaxf(mrun[r], t);
      float alpha = exp2f(mrun[r] - nm);
      mrun[r] = nm;
      float rs = 0.f;
#pragma unroll
      for (int sub = 0; sub < 4; ++sub) {
        float p = exp2f(s[sub][r] - nm);
        s[sub][r] = p;
        rs += p;
      }
      rs += __shfl_xor(rs, 1);
      rs += __shfl_xor(rs, 2);
      rs += __shfl_xor(rs, 4);
      rs += __shfl_xor(rs, 8);
      lrun[r] = lrun[r] * alpha + rs;
#pragma unroll
      for (int c = 0; c < 4; ++c) o[c][r] *= alpha;
    }
    // P (C layout) -> per-wave LDS (16x64 f16 row-major) -> A fragments
#pragma unroll
    for (int r = 0; r < 8; ++r) {
      int m = r + 8 * half;
#pragma unroll
      for (int sub = 0; sub < 4; ++sub)
        pw[m * 64 + sub * 16 + ln] = (_Float16)s[sub][r];
    }
    __builtin_amdgcn_wave_barrier();
    asm volatile("s_wait_dscnt 0" ::: "memory");
    const _Float16* pr = pw + ln * 64;
    v16h pa0 = cat16(*(const h8*)(pr + 16 * half),
                     *(const h8*)(pr + 8 + 16 * half));
    v16h pa1 = cat16(*(const h8*)(pr + 32 + 16 * half),
                     *(const h8*)(pr + 40 + 16 * half));
#pragma unroll
    for (int c = 0; c < 4; ++c) {
      const _Float16* vrow = ldsV + (c * 16 + ln) * 64 + 16 * half;
      v16h bf0 = *(const v16h*)(vrow);       // keys 16h..16h+15
      v16h bf1 = *(const v16h*)(vrow + 32);  // keys 32+16h..
      o[c] = wmma_f16(pa0, bf0, o[c]);
      o[c] = wmma_f16(pa1, bf1, o[c]);
    }
  }
  // finalize: divide by running denom, store f16 (b, n, h*hd)
#pragma unroll
  for (int r = 0; r < 8; ++r) {
    float inv = 1.f / lrun[r];
#pragma unroll
    for (int c = 0; c < 4; ++c) o[c][r] *= inv;
  }
#pragma unroll
  for (int c = 0; c < 4; ++c) {
#pragma unroll
    for (int r = 0; r < 8; ++r) {
      int row = qrow0 + r + 8 * half;
      int col = h * HDIM + c * 16 + ln;
      attnb[((size_t)b * NSEQ + row) * HND + col] = (_Float16)o[c][r];
    }
  }
}

// ---------------------------------------------------------------------------
// Kernel 4: output projection + bias + mask zeroing -> fp32 d_out
// ---------------------------------------------------------------------------
__global__ __launch_bounds__(256) void out_proj_kernel(
    const _Float16* __restrict__ attnb, const float* __restrict__ Wo,
    const float* __restrict__ bo, const unsigned char* __restrict__ maskp,
    float* __restrict__ out) {
  __shared__ __align__(32) _Float16 ldsW[64 * 32];  // col-major Wo tile, 4 KB
  const int tid  = threadIdx.x;
  const int wave = tid >> 5, lane = tid & 31;
  const int half = lane >> 4, ln = lane & 15;
  const int nblk = NSEQ / 64, dblk = DMODEL / 64;
  const int b  = blockIdx.x / (nblk * dblk);
  const int nb = (blockIdx.x / dblk) % nblk;
  const int db = blockIdx.x % dblk;
  const int n0 = nb * 64, d0 = db * 64;
  const int rowg = wave & 3, cg0 = wave >> 2;
  const int arow = n0 + rowg * 16 + ln;

  v8f zero = {};
  v8f acc[2];
  acc[0] = zero; acc[1] = zero;

  for (int kk = 0; kk < HND; kk += 32) {
    __syncthreads();
#pragma unroll
    for (int it = 0; it < 2; ++it) {
      int idx = tid * 2 + it;
      int k   = idx >> 4;
      int n4  = (idx & 15) * 4;
      float4 w4 = *(const float4*)(Wo + (size_t)(kk + k) * DMODEL + d0 + n4);
      ldsW[(n4 + 0) * 32 + k] = (_Float16)w4.x;
      ldsW[(n4 + 1) * 32 + k] = (_Float16)w4.y;
      ldsW[(n4 + 2) * 32 + k] = (_Float16)w4.z;
      ldsW[(n4 + 3) * 32 + k] = (_Float16)w4.w;
    }
    __syncthreads();
    const _Float16* ab = attnb + ((size_t)b * NSEQ + arow) * HND + kk + 16 * half;
    v16h afrag = cat16(*(const h8*)ab, *(const h8*)(ab + 8));
#pragma unroll
    for (int cgi = 0; cgi < 2; ++cgi) {
      const _Float16* bb = ldsW + ((cg0 + cgi * 2) * 16 + ln) * 32 + 16 * half;
      v16h bfrag = *(const v16h*)bb;
      acc[cgi] = wmma_f16(afrag, bfrag, acc[cgi]);
    }
  }
#pragma unroll
  for (int cgi = 0; cgi < 2; ++cgi) {
#pragma unroll
    for (int r = 0; r < 8; ++r) {
      int row = n0 + rowg * 16 + r + 8 * half;
      int col = d0 + (cg0 + cgi * 2) * 16 + ln;
      float val = acc[cgi][r] + bo[col];
      if (!maskp[b * NSEQ + row]) val = 0.f;
      out[((size_t)b * NSEQ + row) * DMODEL + col] = val;
    }
  }
}

// ---------------------------------------------------------------------------
extern "C" void kernel_launch(void* const* d_in, const int* in_sizes, int n_in,
                              void* d_out, int out_size, void* d_ws,
                              size_t ws_size, hipStream_t stream) {
  const float* x            = (const float*)d_in[0];
  const unsigned char* mask = (const unsigned char*)d_in[1];
  const float* freqs        = (const float*)d_in[2];
  const float* Wq = (const float*)d_in[3];
  const float* bq = (const float*)d_in[4];
  const float* Wk = (const float*)d_in[5];
  const float* bk = (const float*)d_in[6];
  const float* Wv = (const float*)d_in[7];
  const float* bv = (const float*)d_in[8];
  const float* qn = (const float*)d_in[9];
  const float* kn = (const float*)d_in[10];
  const float* Wo = (const float*)d_in[11];
  const float* bo = (const float*)d_in[12];

  char* ws = (char*)d_ws;
  const size_t SEG = (size_t)BATCH * NSEQ * DMODEL * sizeof(_Float16);  // 8 MiB
  _Float16* xh    = (_Float16*)(ws + 0 * SEG);
  _Float16* qf    = (_Float16*)(ws + 1 * SEG);
  _Float16* kf    = (_Float16*)(ws + 2 * SEG);
  _Float16* vf    = (_Float16*)(ws + 3 * SEG);
  _Float16* attnb = (_Float16*)(ws + 4 * SEG);

  const int n4 = BATCH * NSEQ * DMODEL / 4;
  cvt_f32_f16<<<1024, 256, 0, stream>>>(x, xh, n4);

  qkv_rope_kernel<<<BATCH * NHEAD * (NSEQ / 64), 256, 0, stream>>>(
      xh, Wq, bq, Wk, bk, Wv, bv, qn, kn, freqs, qf, kf, vf);

  flash_attn_kernel<<<BATCH * NHEAD * (NSEQ / 128), 256, 0, stream>>>(
      qf, kf, vf, mask, attnb);

  out_proj_kernel<<<BATCH * (NSEQ / 64) * (DMODEL / 64), 256, 0, stream>>>(
      attnb, Wo, bo, mask, (float*)d_out);
}
